// DenseGATConv_48747878809940
// MI455X (gfx1250) — matrix operand order repaired and checked by reading, not compiled, not run on hardware
//
#include <hip/hip_runtime.h>

typedef __attribute__((ext_vector_type(16))) _Float16 v16h;
typedef __attribute__((ext_vector_type(8)))  float    v8f;
typedef __attribute__((ext_vector_type(4)))  float    f4;

#define Bb 4
#define Nn 2048
#define Ff 128
#define Hh 4
#define Cc 32
#define HC 128
#define NEG_SLOPE 0.2f
#define CHUNK 256

union AOp { v16h v; f4 q[2]; };

// ---------------------------------------------------------------------------
// Kernel 1: projection xh = x@W (WMMA f16), store xhT (f16, [b][h][c][node]),
// and a_src/a_dst = <xh, att_src/att_dst>.
// grid: (B*N/16) blocks, 128 threads (4 waves; wave w owns head w's 32 cols)
// ---------------------------------------------------------------------------
__global__ __launch_bounds__(128) void gat_proj(
    const float* __restrict__ x, const float* __restrict__ W,
    const float* __restrict__ att_src, const float* __restrict__ att_dst,
    _Float16* __restrict__ xhT, float* __restrict__ a_src, float* __restrict__ a_dst)
{
  __shared__ __align__(16) _Float16 WT[HC][Ff];   // W transposed, f16 (32KB)
  __shared__ __align__(16) _Float16 xs[16][Ff];   // x tile, f16 (4KB)
  __shared__ float xh_s[16][HC];                  // f32 result tile (8KB)
  __shared__ float att_s[Hh][Cc], att_d[Hh][Cc];

  const int tid = threadIdx.x;
  const int m0  = blockIdx.x * 16;      // global row (b*N + n)
  const int b   = m0 / Nn;
  const int n0  = m0 % Nn;

  for (int idx = tid; idx < Ff * HC; idx += 128) {
    int k = idx >> 7, n = idx & 127;
    WT[n][k] = (_Float16)W[idx];
  }
  for (int idx = tid; idx < 16 * Ff; idx += 128) {
    int i = idx >> 7, k = idx & 127;
    xs[i][k] = (_Float16)x[(size_t)(m0 + i) * Ff + k];
  }
  for (int idx = tid; idx < Hh * Cc; idx += 128) {
    att_s[idx >> 5][idx & 31] = att_src[idx];
    att_d[idx >> 5][idx & 31] = att_dst[idx];
  }
  __syncthreads();

  const int wave  = tid >> 5;
  const int lane  = tid & 31;
  const int lr    = lane & 15;
  const int lh    = lane >> 4;
  const int cbase = wave * 32;

  v8f acc0 = {}, acc1 = {};
  #pragma unroll
  for (int ks = 0; ks < 4; ++ks) {
    AOp A, B0, B1;
    A.q[0] = *(const f4*)&xs[lr][ks * 32 + lh * 16];
    A.q[1] = *(const f4*)&xs[lr][ks * 32 + lh * 16 + 8];
    const f4* bp0 = (const f4*)&WT[cbase + lr][ks * 32 + lh * 16];
    const f4* bp1 = (const f4*)&WT[cbase + 16 + lr][ks * 32 + lh * 16];
    B0.q[0] = bp0[0]; B0.q[1] = bp0[1];
    B1.q[0] = bp1[0]; B1.q[1] = bp1[1];
    acc0 = __builtin_amdgcn_wmma_f32_16x16x32_f16(false, A.v, false, B0.v,
                                                  (short)0, acc0, false, false);
    acc1 = __builtin_amdgcn_wmma_f32_16x16x32_f16(false, A.v, false, B1.v,
                                                  (short)0, acc1, false, false);
  }

  // Scatter C tiles: xh_s (f32, for att dots) + xhT (f16, transposed, global)
  #pragma unroll
  for (int a = 0; a < 2; ++a) {
    v8f acc = a ? acc1 : acc0;
    int col = cbase + a * 16 + lr;           // hc column
    int h = col >> 5, c = col & 31;
    union { _Float16 h8[8]; f4 q; } pk;
    #pragma unroll
    for (int r = 0; r < 8; ++r) {
      float v = acc[r];
      xh_s[lh * 8 + r][col] = v;
      pk.h8[r] = (_Float16)v;
    }
    *(f4*)&xhT[((size_t)(b * Hh + h) * Cc + c) * Nn + n0 + lh * 8] = pk.q;
  }
  __syncthreads();

  if (tid < 64) {
    int i = tid >> 2, h = tid & 3;
    float s = 0.f, d = 0.f;
    #pragma unroll
    for (int c = 0; c < Cc; ++c) {
      float v = xh_s[i][h * Cc + c];
      s += v * att_s[h][c];
      d += v * att_d[h][c];
    }
    a_src[(size_t)(m0 + i) * Hh + h] = s;
    a_dst[(size_t)(m0 + i) * Hh + h] = d;
  }
}

// ---------------------------------------------------------------------------
// Kernel 2: fused attention + aggregation, single kernel, alpha written ONCE.
//   Phase A: stream adj once (NT + prefetch), cache edge mask as bytes in LDS,
//            accumulate exp row-sums.
//   Phase B: recompute exp from the LDS mask, normalize, write final alpha
//            (float4 = 4 heads, coalesced, NT), stage normalized f16 weights
//            in LDS and WMMA-aggregate against xhT.
// grid: B*(N/16) blocks, 256 threads (8 waves; wave -> (head, c-half)).
// ---------------------------------------------------------------------------
__global__ __launch_bounds__(256) void gat_attn(
    const int* __restrict__ adj, const _Float16* __restrict__ xhT,
    const float* __restrict__ a_src, const float* __restrict__ a_dst,
    const float* __restrict__ bias,
    float* __restrict__ out, float* __restrict__ alpha)
{
  __shared__ f4 asrc_s[Nn];                               // 32KB
  __shared__ f4 adst_s[16];
  __shared__ unsigned char ebuf[16][Nn];                  // 32KB edge mask
  __shared__ __align__(16) _Float16 wbuf[Hh][16][CHUNK];  // 32KB
  __shared__ float red[16][Hh][16];                       // 4KB
  __shared__ f4 invs_s[16];

  const int tid = threadIdx.x;
  const int b   = blockIdx.x >> 7;
  const int i0  = (blockIdx.x & 127) * 16;

  const f4* asg = (const f4*)a_src + (size_t)b * Nn;
  for (int idx = tid; idx < Nn; idx += 256) asrc_s[idx] = asg[idx];
  if (tid < 16) adst_s[tid] = ((const f4*)a_dst)[(size_t)b * Nn + i0 + tid];
  __syncthreads();

  const int iC   = tid >> 4;        // row within tile (compute phases)
  const int js   = tid & 15;        // j sub-lane (compute phases)
  const int lane = tid & 31;
  const int lr   = lane & 15;
  const int lh   = lane >> 4;
  const int wave = tid >> 5;
  const int h_w  = wave >> 1;       // head for this wave
  const int coff = (wave & 1) * 16; // c-half

  const f4 ad = adst_s[iC];
  const size_t arow = (size_t)(b * Nn + i0 + iC) * Nn;
  f4* alpha4 = (f4*)alpha;
  const _Float16* xb = xhT + ((size_t)(b * Hh + h_w) * Cc + coff + lr) * Nn;

  // ---------------- Phase A: edge mask + exp row-sums ----------------
  float sumh[4] = {0.f, 0.f, 0.f, 0.f};
  #pragma unroll 4
  for (int k = 0; k < Nn / 16; ++k) {
    int j = js + k * 16;
    if ((k & 15) == 0)   // sparse prefetch ~4KB ahead (global_prefetch_b8)
      __builtin_prefetch(&adj[arow + j + 1024], 0, 1);
    int av = __builtin_nontemporal_load(&adj[arow + j]);
    bool edge = (av != 0) || (j == i0 + iC);   // self loop
    ebuf[iC][j] = edge ? 1 : 0;
    f4 aj = asrc_s[j];
    #pragma unroll
    for (int h = 0; h < 4; ++h) {
      float e = aj[h] + ad[h];
      e = e > 0.f ? e : NEG_SLOPE * e;
      sumh[h] += edge ? __expf(e) : 0.f;
    }
  }
  #pragma unroll
  for (int h = 0; h < 4; ++h) red[iC][h][js] = sumh[h];
  __syncthreads();
  if (tid < 64) {
    int i = tid >> 2, h = tid & 3;
    float t = 0.f;
    #pragma unroll
    for (int k = 0; k < 16; ++k) t += red[i][h][k];
    invs_s[i][h] = 1.f / t;
  }
  __syncthreads();

  // ---------------- Phase B: normalized alpha + WMMA aggregation ----------
  const f4 iv = invs_s[iC];
  v8f acc = {};
  for (int jc = 0; jc < Nn; jc += CHUNK) {
    #pragma unroll 4
    for (int k = 0; k < 16; ++k) {
      int j = jc + js + k * 16;
      bool edge = ebuf[iC][j] != 0;
      f4 aj = asrc_s[j];
      f4 wv;
      #pragma unroll
      for (int h = 0; h < 4; ++h) {
        float e = aj[h] + ad[h];
        e = e > 0.f ? e : NEG_SLOPE * e;
        float w = edge ? __expf(e) * iv[h] : 0.f;   // normalized
        wv[h] = w;
        wbuf[h][iC][j - jc] = (_Float16)w;
      }
      __builtin_nontemporal_store(wv, &alpha4[arow + j]);  // final alpha
    }
    __syncthreads();
    #pragma unroll
    for (int ks = 0; ks < CHUNK / 32; ++ks) {
      AOp A, Bv;
      A.q[0] = *(const f4*)&wbuf[h_w][lr][ks * 32 + lh * 16];
      A.q[1] = *(const f4*)&wbuf[h_w][lr][ks * 32 + lh * 16 + 8];
      const f4* bp = (const f4*)&xb[jc + ks * 32 + lh * 16];
      Bv.q[0] = bp[0]; Bv.q[1] = bp[1];
      acc = __builtin_amdgcn_wmma_f32_16x16x32_f16(false, A.v, false, Bv.v,
                                                   (short)0, acc, false, false);
    }
    __syncthreads();
  }

  // ---- final out = acc + bias (weights already normalized) ----
  #pragma unroll
  for (int r = 0; r < 8; ++r) {
    int i = lh * 8 + r;
    int col = h_w * Cc + coff + lr;
    out[(size_t)(b * Nn + i0 + i) * HC + col] = acc[r] + bias[col];
  }
}

extern "C" void kernel_launch(void* const* d_in, const int* in_sizes, int n_in,
                              void* d_out, int out_size, void* d_ws, size_t ws_size,
                              hipStream_t stream) {
  (void)in_sizes; (void)n_in; (void)out_size; (void)ws_size;
  const float* x       = (const float*)d_in[0];
  const int*   adj     = (const int*)  d_in[1];
  const float* W       = (const float*)d_in[2];
  const float* att_src = (const float*)d_in[3];
  const float* att_dst = (const float*)d_in[4];
  const float* bias    = (const float*)d_in[5];

  float* out   = (float*)d_out;
  float* alpha = out + (size_t)Bb * Nn * HC;   // outputs concatenated: out, alpha

  char* ws = (char*)d_ws;
  _Float16* xhT = (_Float16*)ws;                               // 2 MB
  float* a_src  = (float*)(ws + (size_t)Bb * Hh * Cc * Nn * 2);
  float* a_dst  = a_src + (size_t)Bb * Nn * Hh;

  gat_proj<<<(Bb * Nn) / 16, 128, 0, stream>>>(x, W, att_src, att_dst,
                                               xhT, a_src, a_dst);
  gat_attn<<<Bb * (Nn / 16), 256, 0, stream>>>(adj, xhT, a_src, a_dst, bias,
                                               out, alpha);
}
